// LaneAttention_30786325578415
// MI455X (gfx1250) — matrix operand order repaired and checked by reading, not compiled, not run on hardware
//
#include <hip/hip_runtime.h>
#include <hip/hip_bf16.h>

typedef __attribute__((ext_vector_type(16))) __bf16         v16bf;
typedef __attribute__((ext_vector_type(8)))  float          v8f;
typedef __attribute__((ext_vector_type(16))) unsigned short v16u;
typedef __attribute__((ext_vector_type(8)))  unsigned int   v8ui;

#define N_OBS   16384
#define M_LANES 131072
#define D_IN    128
#define D_OUT   512

__device__ __forceinline__ unsigned short f32_to_bf16(float x) {
    unsigned int u = __float_as_uint(x);
    u += 0x7FFFu + ((u >> 16) & 1u);          // round-to-nearest-even
    return (unsigned short)(u >> 16);
}

// ---------------------------------------------------------------- init
__global__ void la_zero_kernel(float* out, int out_n, int* cnt, int cnt_n) {
    int idx = blockIdx.x * blockDim.x + threadIdx.x;
    int stride = gridDim.x * blockDim.x;
    for (int i = idx; i < out_n; i += stride) out[i] = 0.0f;
    for (int i = idx; i < cnt_n; i += stride) cnt[i] = 0;
}

// ------------------------------------------------- pack W into B-fragments
// fragment f = kb*32+nb (kb: K-block of 32, nb: N-block of 16)
// per lane l: 16 bf16 = VGPR pairs; col = nb*16 + l%16, K = kb*32 + (l/16)*16 + 2g + j
__global__ void la_packW_kernel(const float* __restrict__ W, unsigned short* __restrict__ wsW) {
    int p = blockIdx.x * blockDim.x + threadIdx.x;   // 4*32*32*16 = 65536
    int e    = p & 15;
    int l    = (p >> 4) & 31;
    int f    = p >> 9;
    int nb   = f & 31;
    int kb   = f >> 5;
    int g    = e >> 1, j = e & 1;
    int half = l >> 4;
    int col  = nb * 16 + (l & 15);
    int K    = kb * 32 + half * 16 + g * 2 + j;
    wsW[p] = f32_to_bf16(W[K * D_OUT + col]);
}

// ---------------------------------------------------------------- counts
__global__ void la_count_kernel(const int* __restrict__ seg, int* cnt) {
    int i = blockIdx.x * blockDim.x + threadIdx.x;
    if (i < M_LANES)
        __hip_atomic_fetch_add(&cnt[seg[i]], 1, __ATOMIC_RELAXED, __HIP_MEMORY_SCOPE_AGENT);
}

// ------------------------------------- fused GEMM + ReLU + segmented reduce
__global__ void __launch_bounds__(256)
la_main_kernel(const float* __restrict__ lane_enc, const int* __restrict__ seg,
               const unsigned short* __restrict__ wsW, const float* __restrict__ bias,
               float* __restrict__ out) {
    __shared__ unsigned short Abf[16 * 128];   // 16 rows x 128 K, bf16
    __shared__ float enc[16 * 512];            // post-ReLU tile
    __shared__ int   segT[16];

    const int tid = threadIdx.x;
    const int m0  = blockIdx.x * 16;

    // stage A tile (f32 -> bf16)
    #pragma unroll
    for (int k = 0; k < 8; ++k) {
        int idx = tid + k * 256;
        int r = idx >> 7, c = idx & 127;
        Abf[idx] = f32_to_bf16(lane_enc[(size_t)(m0 + r) * D_IN + c]);
    }
    if (tid < 16) segT[tid] = seg[m0 + tid];
    __syncthreads();

    const int wave   = tid >> 5;       // 0..7, owns N columns [wave*64, wave*64+64)
    const int lane32 = tid & 31;
    const int half   = lane32 >> 4;
    const int l16    = lane32 & 15;    // A row / C column within fragment

    __builtin_prefetch(&wsW[(size_t)(wave * 4) * 32 * 16], 0, 1);   // global_prefetch

    v8f acc[4] = {};                   // 4 N-fragments of 16x16 f32

    #pragma unroll
    for (int kb = 0; kb < 4; ++kb) {
        // A fragment: contiguous K per lane half (K = half*16 + s), matching packed B
        v8ui au;
        #pragma unroll
        for (int g = 0; g < 8; ++g) {
            int Kk = kb * 32 + half * 16 + g * 2;
            au[g] = *(const unsigned int*)&Abf[l16 * 128 + Kk];
        }
        v16bf a = __builtin_bit_cast(v16bf, au);

        #pragma unroll
        for (int nf = 0; nf < 4; ++nf) {
            int nb = wave * 4 + nf;
            const v16u* bp = (const v16u*)&wsW[((size_t)(kb * 32 + nb) * 32 + lane32) * 16];
            v16bf b = __builtin_bit_cast(v16bf, *bp);
            acc[nf] = __builtin_amdgcn_wmma_f32_16x16x32_bf16(
                false, a, false, b, (short)0, acc[nf], false, false);
        }
    }

    // bias + ReLU, scatter C fragments to LDS (VGPR r -> M = r + 8*half, N = l16)
    #pragma unroll
    for (int nf = 0; nf < 4; ++nf) {
        int nc = (wave * 4 + nf) * 16 + l16;
        float bv = bias[nc];
        #pragma unroll
        for (int r = 0; r < 8; ++r) {
            int mrow = half * 8 + r;
            float v = acc[nf][r] + bv;
            enc[mrow * 512 + nc] = v > 0.0f ? v : 0.0f;
        }
    }
    __syncthreads();

    // run-length segmented reduce over the sorted seg ids of this 16-row tile;
    // one atomic pair per (segment-run, column). ReLU >= 0 so uint-max == float-max.
    #pragma unroll
    for (int cc = 0; cc < 2; ++cc) {
        int col = tid + cc * 256;
        int curSeg = segT[0];
        float mx = 0.0f, sm = 0.0f;
        #pragma unroll
        for (int r = 0; r < 16; ++r) {
            int s = segT[r];
            float v = enc[r * 512 + col];
            if (s != curSeg) {
                __hip_atomic_fetch_max((unsigned int*)&out[curSeg * 1024 + col],
                                       __float_as_uint(mx),
                                       __ATOMIC_RELAXED, __HIP_MEMORY_SCOPE_AGENT);
                __hip_atomic_fetch_add(&out[curSeg * 1024 + 512 + col], sm,
                                       __ATOMIC_RELAXED, __HIP_MEMORY_SCOPE_AGENT);
                curSeg = s; mx = 0.0f; sm = 0.0f;
            }
            mx = fmaxf(mx, v);
            sm += v;
        }
        __hip_atomic_fetch_max((unsigned int*)&out[curSeg * 1024 + col],
                               __float_as_uint(mx),
                               __ATOMIC_RELAXED, __HIP_MEMORY_SCOPE_AGENT);
        __hip_atomic_fetch_add(&out[curSeg * 1024 + 512 + col], sm,
                               __ATOMIC_RELAXED, __HIP_MEMORY_SCOPE_AGENT);
    }
}

// ---------------------------------------------------------------- finalize
__global__ void la_finalize_kernel(float* out, const int* __restrict__ cnt) {
    int i = blockIdx.x * blockDim.x + threadIdx.x;  // N_OBS*512
    int row = i >> 9, c = i & 511;
    out[row * 1024 + 512 + c] *= (1.0f / (float)cnt[row]);
}

extern "C" void kernel_launch(void* const* d_in, const int* in_sizes, int n_in,
                              void* d_out, int out_size, void* d_ws, size_t ws_size,
                              hipStream_t stream) {
    (void)in_sizes; (void)n_in; (void)ws_size;
    const float* lane = (const float*)d_in[1];   // lane_encoding (131072,128)
    const int*   seg  = (const int*)d_in[2];     // same_obs_mask (131072,1)
    const float* W    = (const float*)d_in[3];   // (128,512)
    const float* bias = (const float*)d_in[4];   // (512,)
    float* out = (float*)d_out;                  // (16384,1024)

    unsigned short* wsW = (unsigned short*)d_ws;                 // 128*512 bf16 = 128 KB
    int* cnt = (int*)((char*)d_ws + (size_t)D_IN * D_OUT * sizeof(unsigned short));

    la_zero_kernel<<<2048, 256, 0, stream>>>(out, out_size, cnt, N_OBS);
    la_packW_kernel<<<(D_IN * D_OUT) / 256, 256, 0, stream>>>(W, wsW);
    la_count_kernel<<<M_LANES / 256, 256, 0, stream>>>(seg, cnt);
    la_main_kernel<<<M_LANES / 16, 256, 0, stream>>>(lane, seg, wsW, bias, out);
    la_finalize_kernel<<<(N_OBS * 512) / 256, 256, 0, stream>>>(out, cnt);
}
